// RoPEFusedMHA_74406013436469
// MI455X (gfx1250) — matrix operand order repaired and checked by reading, not compiled, not run on hardware
//
#include <hip/hip_runtime.h>
#include <hip/hip_bf16.h>

// ---------------------------------------------------------------------------
// RoPE + fused MHA for MI455X (gfx1250, wave32, WMMA bf16 16x16x32).
// v3: fixed async-load builtin pointer types (AS1/AS3 i32x4*). LDS-staged
// operands via CDNA5 async/TDM paths, double-buffered K/V, software-
// pipelined weight stream. fp32 in/out, bf16 compute, f32 accum.
// ---------------------------------------------------------------------------

typedef __bf16 bf16;
typedef __attribute__((ext_vector_type(8)))  __bf16       v8bf;
typedef __attribute__((ext_vector_type(16))) __bf16       v16bf;
typedef __attribute__((ext_vector_type(8)))  float        v8f;
typedef __attribute__((ext_vector_type(4)))  unsigned int u32x4;
typedef __attribute__((ext_vector_type(4)))  int          i32x4;
typedef __attribute__((ext_vector_type(8)))  int          i32x8;

#define WMMA_BF16(a, b, c) \
  __builtin_amdgcn_wmma_f32_16x16x32_bf16(false, (a), false, (b), (short)0, (c), false, false)
#define CAT16(lo, hi) __builtin_shufflevector((lo), (hi), 0,1,2,3,4,5,6,7,8,9,10,11,12,13,14,15)

constexpr int BB = 2, SS = 2048, DMm = 1024, HH = 16, DH = 64;
constexpr float LOG_THETA_OVER_HALF_D = 0.28782313662425575f; // ln(10000)/32

// ---- gfx1250 async / TDM feature probes ------------------------------------
#if __has_builtin(__builtin_amdgcn_global_load_async_to_lds_b128)
#  define HAS_ASYNC_LDS 1
#endif
#if __has_builtin(__builtin_amdgcn_tensor_load_to_lds)
#  define HAS_TDM 1
#  if __has_include(<hip/amd_detail/amd_gfx1250_TDM.h>)
#    define TDM_ARITY 6   /* amdgpu-toolchain / therock-10.0 headers */
#  else
#    define TDM_ARITY 5   /* ROCm 7.2 clang-22 */
#  endif
#endif

#if __has_builtin(__builtin_amdgcn_s_wait_asynccnt)
#  define WAIT_ASYNC(n) __builtin_amdgcn_s_wait_asynccnt(n)
#elif defined(HAS_ASYNC_LDS)
#  define WAIT_ASYNC(n) asm volatile("s_wait_asynccnt %0" ::"i"(n) : "memory")
#else
#  define WAIT_ASYNC(n) ((void)0)
#endif

#define LDS_AS(p) ((__attribute__((address_space(3))) void*)(p))
// Pointer casts for the async builtin: AS1/AS3 pointers to 4-dword vectors.
#define GLB_V4(p) ((__attribute__((address_space(1))) i32x4*)(unsigned long long)(p))
#define LDS_V4(p) ((__attribute__((address_space(3))) i32x4*)(p))

// 16-byte global -> LDS stage (async if available).
__device__ __forceinline__ void stage16(bf16* ldst, const bf16* gsrc) {
#ifdef HAS_ASYNC_LDS
  __builtin_amdgcn_global_load_async_to_lds_b128(GLB_V4(gsrc), LDS_V4(ldst), 0, 0);
#else
  *(v8bf*)ldst = *(const v8bf*)gsrc;
#endif
}

#ifdef HAS_TDM
// Contiguous n-element (bf16) global -> LDS copy through the Tensor Data Mover.
// 2D descriptor degenerated to a single row: tile_dim0 = tensor_dim0 = n.
__device__ __forceinline__ void tdm_copy_bf16(bf16* lds_dst, const bf16* gsrc, unsigned n) {
  unsigned long long ga = (unsigned long long)gsrc;
  unsigned lds_off = (unsigned)(unsigned long long)LDS_AS(lds_dst);
  u32x4 g0 = {0, 0, 0, 0};
  i32x8 g1 = {0, 0, 0, 0, 0, 0, 0, 0};
  i32x4 g2 = {0, 0, 0, 0};
  i32x4 g3 = {0, 0, 0, 0};
  g0[0] = 1u;                                       // count = 1 (valid, user mode)
  g0[1] = lds_off;                                  // lds_addr
  g0[2] = (unsigned)ga;                             // global_addr[31:0]
  g0[3] = (unsigned)(ga >> 32) | (2u << 30);        // global_addr[56:32] | type=2
  g1[0] = (int)(1u << 16);                          // data_size = 1 (2 bytes)
  g1[1] = (int)((n & 0xFFFFu) << 16);               // tensor_dim0[15:0]  (bits 63:48)
  g1[2] = (int)((n >> 16) | (1u << 16));            // tensor_dim0[31:16] | tensor_dim1 = 1
  g1[3] = (int)(n << 16);                           // tile_dim0 = n      (bits 127:112)
  g1[4] = 1;                                        // tile_dim1 = 1
  g1[5] = (int)n;                                   // tensor_dim0_stride[31:0]
#if TDM_ARITY == 6
  i32x8 g4 = {0, 0, 0, 0, 0, 0, 0, 0};
  __builtin_amdgcn_tensor_load_to_lds(g0, g1, g2, g3, g4, 0);
#else
  __builtin_amdgcn_tensor_load_to_lds(g0, g1, g2, g3, 0);
#endif
}
#endif

// ---------------------------------------------------------------------------
// Pass 0a: fp32 -> bf16 flat convert
__global__ void k_cvt_bf16(const float* __restrict__ src, bf16* __restrict__ dst, int n) {
  int i = blockIdx.x * blockDim.x + threadIdx.x;
  if (i < n) dst[i] = (bf16)src[i];
}

// Pass 0b: fp32 (K x N) -> bf16 (N x K) transpose-convert (K = N = 1024)
__global__ void k_cvt_t(const float* __restrict__ src, bf16* __restrict__ dst) {
  int i = blockIdx.x * blockDim.x + threadIdx.x;
  int k = i >> 10, n = i & 1023;
  dst[(size_t)n * DMm + k] = (bf16)src[i];
}

// ---------------------------------------------------------------------------
// Stage the block-shared 16x1024 bf16 A panel (32 KB, contiguous) into LDS.
__device__ __forceinline__ void stage_a_panel(bf16* atile, const bf16* asrc,
                                              int tid, int wid) {
#ifdef HAS_TDM
  if (wid == 0) {
    tdm_copy_bf16(atile, asrc, 16 * DMm);
    __builtin_amdgcn_s_wait_tensorcnt(0);
  }
#else
#pragma unroll
  for (int i = 0; i < 8; i++)
    stage16(atile + tid * 8 + i * 2048, asrc + tid * 8 + i * 2048);
  WAIT_ASYNC(0);
#endif
  __syncthreads();
}

// ---------------------------------------------------------------------------
// Pass 1: QKV projection + RoPE.  Grid (rowTile=256, group=6), 8 waves/block.
// The 8 waves of a block share one LDS-staged A panel and each own one
// (z, head) column strip of 64: 4 accumulators, K-loop of 32 with a
// register-double-buffered weight stream.
__global__ __launch_bounds__(256) void k_qkv_rope(
    const bf16* __restrict__ hid, const bf16* __restrict__ wq_t,
    const bf16* __restrict__ wk_t, const bf16* __restrict__ wv_t,
    const int* __restrict__ pos_ids,
    bf16* __restrict__ q_ws, bf16* __restrict__ k_ws, bf16* __restrict__ v_ws) {
  __shared__ __align__(64) bf16 atile[16 * DMm];
  const int tid  = threadIdx.x;
  const int lane = tid & 31;
  const int wid  = tid >> 5;
  const int rowTile = blockIdx.x;                // 0..255
  const int zh   = blockIdx.y * 8 + wid;         // 0..47
  const int z    = zh >> 4;                      // 0=q 1=k 2=v
  const int head = zh & 15;
  const int half = lane >> 4;
  const int nl   = lane & 15;

  stage_a_panel(atile, hid + (size_t)rowTile * 16 * DMm, tid, wid);

  const bf16* wt  = (z == 0) ? wq_t : (z == 1) ? wk_t : wv_t;
  const bf16* wb0 = wt + (size_t)(head * 64 + nl) * DMm + half * 16; // B col = lane&15
  const bf16* ar  = atile + nl * DMm;                                // A row = lane&15

  v8f acc0 = {}, acc1 = {}, acc2 = {}, acc3 = {};
  v16bf bc0 = *(const v16bf*)(wb0);
  v16bf bc1 = *(const v16bf*)(wb0 + 16 * DMm);
  v16bf bc2 = *(const v16bf*)(wb0 + 32 * DMm);
  v16bf bc3 = *(const v16bf*)(wb0 + 48 * DMm);
  for (int kk = 0; kk < DMm; kk += 32) {
    v16bf bn0 = bc0, bn1 = bc1, bn2 = bc2, bn3 = bc3;
    if (kk + 32 < DMm) {                         // prefetch next K-slab into regs
      const bf16* wb = wb0 + kk + 32;
      bn0 = *(const v16bf*)(wb);
      bn1 = *(const v16bf*)(wb + 16 * DMm);
      bn2 = *(const v16bf*)(wb + 32 * DMm);
      bn3 = *(const v16bf*)(wb + 48 * DMm);
      __builtin_prefetch(wb + 64, 0, 1);         // global_prefetch_b8
    }
    v16bf a = CAT16(*(const v8bf*)(ar + kk + half * 16),
                    *(const v8bf*)(ar + kk + half * 16 + 8));   // ds_load_b128 x2
    acc0 = WMMA_BF16(a, bc0, acc0);
    acc1 = WMMA_BF16(a, bc1, acc1);
    acc2 = WMMA_BF16(a, bc2, acc2);
    acc3 = WMMA_BF16(a, bc3, acc3);
    bc0 = bn0; bc1 = bn1; bc2 = bn2; bc3 = bn3;
  }

  if (z < 2) {
    // RoPE: columns of acc_t are d = t*16 + nl; rotate pairs (d, d+32).
    const float inv0 = __expf(-(float)nl * LOG_THETA_OVER_HALF_D);
    const float inv1 = __expf(-(float)(16 + nl) * LOG_THETA_OVER_HALF_D);
    bf16* outp = (z == 0) ? q_ws : k_ws;
#pragma unroll
    for (int j = 0; j < 8; j++) {
      const int r  = rowTile * 16 + j + 8 * half;       // C row = j + 8*half
      const int bb = r >> 11, ss = r & (SS - 1);
      const float pos = (float)pos_ids[bb * SS + ss];
      const float a0 = pos * inv0, a1 = pos * inv1;
      const float c0 = __cosf(a0), s0 = __sinf(a0);
      const float c1 = __cosf(a1), s1 = __sinf(a1);
      const float x0 = acc0[j], x1 = acc1[j], x2 = acc2[j], x3 = acc3[j];
      const size_t base = ((size_t)(bb * HH + head) * SS + ss) * DH;
      outp[base + 0 * 16 + nl] = (bf16)(x0 * c0 - x2 * s0);
      outp[base + 1 * 16 + nl] = (bf16)(x1 * c1 - x3 * s1);
      outp[base + 2 * 16 + nl] = (bf16)(x2 * c0 + x0 * s0);
      outp[base + 3 * 16 + nl] = (bf16)(x3 * c1 + x1 * s1);
    }
  } else {
    // V stored transposed: (B,H,D,S)
#pragma unroll
    for (int j = 0; j < 8; j++) {
      const int r  = rowTile * 16 + j + 8 * half;
      const int bb = r >> 11, ss = r & (SS - 1);
      const size_t base = (size_t)(bb * HH + head) * DH * SS + ss;
      v_ws[base + (size_t)(0 * 16 + nl) * SS] = (bf16)acc0[j];
      v_ws[base + (size_t)(1 * 16 + nl) * SS] = (bf16)acc1[j];
      v_ws[base + (size_t)(2 * 16 + nl) * SS] = (bf16)acc2[j];
      v_ws[base + (size_t)(3 * 16 + nl) * SS] = (bf16)acc3[j];
    }
  }
}

// ---------------------------------------------------------------------------
// Pass 2: flash attention.  Grid (bh=32, qgroup=16), 8 waves/block, all waves
// share (b,h).  Per 32-key step the block stages K (32x64) and V^T (64x32)
// chunks into a double-buffered LDS pipeline via async loads; each wave does
// 4 score WMMAs, online softmax (shfl_xor row butterflies), a wave-private
// LDS bounce of P, then 4 ctx WMMAs with the rescaled accumulator as C.
__global__ __launch_bounds__(256) void k_attn(
    const bf16* __restrict__ q_ws, const bf16* __restrict__ k_ws,
    const bf16* __restrict__ v_ws, const float* __restrict__ mask,
    bf16* __restrict__ ctx_ws) {
  __shared__ __align__(64) bf16 kbuf[2][32 * 64];   // (key, d)
  __shared__ __align__(64) bf16 vbuf[2][32 * 64];   // (d, key)
  __shared__ __align__(64) bf16 pbuf[8][16 * 32];   // per-wave P tile
  const int tid  = threadIdx.x;
  const int lane = tid & 31;
  const int wid  = tid >> 5;
  const int bh   = blockIdx.x;                 // 0..31
  const int qt   = blockIdx.y * 8 + wid;       // 0..127
  const int b    = bh >> 4;
  const int h    = bh & 15;
  const int half = lane >> 4;
  const int nl   = lane & 15;

  const bf16* ksrc = k_ws + (size_t)bh * SS * DH;   // chunk jj: contiguous 2048 elems
  const bf16* vsrc = v_ws + (size_t)bh * DH * SS;   // row d: 32 contiguous keys
  const int kofs = tid * 8;                         // this thread's 16B K slice
  const int vd   = tid >> 2;                        // V row (d)
  const int vseg = (tid & 3) * 8;                   // V key segment

  auto stage = [&](int buf, int jj) {
    stage16(&kbuf[buf][kofs], ksrc + (size_t)jj * 2048 + kofs);
    stage16(&vbuf[buf][vd * 32 + vseg], vsrc + (size_t)vd * SS + jj * 32 + vseg);
  };

  // Q A-fragments (row = lane&15), d split 0..31 / 32..63, loop-resident.
  const bf16* qbase = q_ws + ((size_t)bh * SS + qt * 16 + nl) * DH;
  const v16bf qa0 = CAT16(*(const v8bf*)(qbase + half * 16),
                          *(const v8bf*)(qbase + half * 16 + 8));
  const v16bf qa1 = CAT16(*(const v8bf*)(qbase + 32 + half * 16),
                          *(const v8bf*)(qbase + 40 + half * 16));

  float rmax[8], rsum[8];
#pragma unroll
  for (int j = 0; j < 8; j++) { rmax[j] = -1e30f; rsum[j] = 0.f; }
  v8f o0 = {}, o1 = {}, o2 = {}, o3 = {};

  const float* mrow = mask + (size_t)b * SS * SS;

  stage(0, 0);
  for (int jj = 0; jj < 64; jj++) {
    const int cur = jj & 1;
    if (jj + 1 < 64) { stage(cur ^ 1, jj + 1); WAIT_ASYNC(2); }
    else             { WAIT_ASYNC(0); }
    __syncthreads();                       // staged chunk visible block-wide

    v8f st[2];
#pragma unroll
    for (int t = 0; t < 2; t++) {
      const bf16* kp = &kbuf[cur][(t * 16 + nl) * 64 + half * 16];
      v16bf bk0 = *(const v16bf*)(kp);         // d rows 0..31
      v16bf bk1 = *(const v16bf*)(kp + 32);    // d rows 32..63
      v8f s_ = {};
      s_ = WMMA_BF16(qa0, bk0, s_);
      s_ = WMMA_BF16(qa1, bk1, s_);
      const int key = jj * 32 + t * 16 + nl;
#pragma unroll
      for (int j = 0; j < 8; j++) {            // additive mask (T5-style)
        const int sq = qt * 16 + j + 8 * half;
        s_[j] += mrow[(size_t)sq * SS + key];
      }
      st[t] = s_;
    }
    // Online softmax. A C-row spans the 16 lanes of one half -> xor butterflies.
    float p0a[8], p1a[8], alpha[8];
#pragma unroll
    for (int j = 0; j < 8; j++) {
      float v = fmaxf(st[0][j], st[1][j]);
      v = fmaxf(v, __shfl_xor(v, 1));
      v = fmaxf(v, __shfl_xor(v, 2));
      v = fmaxf(v, __shfl_xor(v, 4));
      v = fmaxf(v, __shfl_xor(v, 8));
      const float nm = fmaxf(rmax[j], v);
      const float al = __expf(rmax[j] - nm);
      const float p0 = __expf(st[0][j] - nm);
      const float p1 = __expf(st[1][j] - nm);
      float ps = p0 + p1;
      ps += __shfl_xor(ps, 1);
      ps += __shfl_xor(ps, 2);
      ps += __shfl_xor(ps, 4);
      ps += __shfl_xor(ps, 8);
      rsum[j] = rsum[j] * al + ps;
      rmax[j] = nm;
      alpha[j] = al; p0a[j] = p0; p1a[j] = p1;
    }
    // Wave-private P bounce through LDS (per-wave DS ordering, no barrier).
#pragma unroll
    for (int j = 0; j < 8; j++) {
      o0[j] *= alpha[j]; o1[j] *= alpha[j]; o2[j] *= alpha[j]; o3[j] *= alpha[j];
      const int mm = j + 8 * half;
      pbuf[wid][mm * 32 + nl]      = (bf16)p0a[j];
      pbuf[wid][mm * 32 + 16 + nl] = (bf16)p1a[j];
    }
    v16bf pa;
#pragma unroll
    for (int e = 0; e < 8; e++) {
      pa[e]     = pbuf[wid][nl * 32 + half * 16 + e];       // K 0..7 / 16..23
      pa[8 + e] = pbuf[wid][nl * 32 + half * 16 + 8 + e];   // K 8..15 / 24..31
    }
    // V^T B-fragments from LDS: column = d, K rows = key (contiguous).
    const bf16* vp = &vbuf[cur][nl * 32 + half * 16];
    v16bf bv0 = *(const v16bf*)(vp);
    v16bf bv1 = *(const v16bf*)(vp + 16 * 32);
    v16bf bv2 = *(const v16bf*)(vp + 32 * 32);
    v16bf bv3 = *(const v16bf*)(vp + 48 * 32);
    o0 = WMMA_BF16(pa, bv0, o0);
    o1 = WMMA_BF16(pa, bv1, o1);
    o2 = WMMA_BF16(pa, bv2, o2);
    o3 = WMMA_BF16(pa, bv3, o3);
    __syncthreads();                       // all reads of `cur` done before reuse
  }
  // Normalize, store ctx as bf16 in (B, S, H*D).
#pragma unroll
  for (int j = 0; j < 8; j++) {
    const float inv = 1.0f / rsum[j];
    const int sq = qt * 16 + j + 8 * half;
    const size_t base = ((size_t)b * SS + sq) * DMm + h * DH;
    ctx_ws[base + 0 * 16 + nl] = (bf16)(o0[j] * inv);
    ctx_ws[base + 1 * 16 + nl] = (bf16)(o1[j] * inv);
    ctx_ws[base + 2 * 16 + nl] = (bf16)(o2[j] * inv);
    ctx_ws[base + 3 * 16 + nl] = (bf16)(o3[j] * inv);
  }
}

// ---------------------------------------------------------------------------
// Pass 3: out = ctx @ Wo.  Grid (rowTile=256, colgroup=8); LDS-staged A panel
// shared by the 8 waves, register-double-buffered Wo stream, fp32 output.
__global__ __launch_bounds__(256) void k_oproj(
    const bf16* __restrict__ ctx, const bf16* __restrict__ wo_t,
    float* __restrict__ out) {
  __shared__ __align__(64) bf16 atile[16 * DMm];
  const int tid  = threadIdx.x;
  const int lane = tid & 31;
  const int wid  = tid >> 5;
  const int rowTile = blockIdx.x;                 // 0..255
  const int colTile = blockIdx.y * 8 + wid;       // 0..63
  const int half = lane >> 4, nl = lane & 15;

  stage_a_panel(atile, ctx + (size_t)rowTile * 16 * DMm, tid, wid);

  const bf16* brow = wo_t + (size_t)(colTile * 16 + nl) * DMm + half * 16;
  const bf16* ar   = atile + nl * DMm;
  v8f acc = {};
  v16bf bc = *(const v16bf*)(brow);
  for (int kk = 0; kk < DMm; kk += 32) {
    v16bf bn = bc;
    if (kk + 32 < DMm) {
      bn = *(const v16bf*)(brow + kk + 32);
      __builtin_prefetch(brow + kk + 64, 0, 1);
    }
    v16bf a = CAT16(*(const v8bf*)(ar + kk + half * 16),
                    *(const v8bf*)(ar + kk + half * 16 + 8));
    acc = WMMA_BF16(a, bc, acc);
    bc = bn;
  }
#pragma unroll
  for (int j = 0; j < 8; j++) {
    const int r = rowTile * 16 + j + 8 * half;
    out[(size_t)r * DMm + colTile * 16 + nl] = acc[j];
  }
}

// ---------------------------------------------------------------------------
extern "C" void kernel_launch(void* const* d_in, const int* in_sizes, int n_in,
                              void* d_out, int out_size, void* d_ws, size_t ws_size,
                              hipStream_t stream) {
  const float* hid  = (const float*)d_in[0];  // (B,S,DM)
  const float* mask = (const float*)d_in[1];  // (B,1,S,S)
  const int*   pos  = (const int*)d_in[2];    // (B,S)
  const float* Wq   = (const float*)d_in[3];
  const float* Wk   = (const float*)d_in[4];
  const float* Wv   = (const float*)d_in[5];
  const float* Wo   = (const float*)d_in[6];

  // Workspace layout (48 MB total):
  //  [0,8M)   hidden bf16        [8M,16M)  Wq^T/Wk^T/Wv^T/Wo^T bf16 (2M each)
  //  [16,24M) Q bf16 (B,H,S,D)   [24,32M)  K bf16 (B,H,S,D)
  //  [32,40M) V bf16 (B,H,D,S)   [40,48M)  ctx bf16 (B,S,H*D)
  char* ws = (char*)d_ws;
  bf16* hid_bf = (bf16*)(ws);
  bf16* wq_t   = (bf16*)(ws + ( 8u << 20));
  bf16* wk_t   = (bf16*)(ws + (10u << 20));
  bf16* wv_t   = (bf16*)(ws + (12u << 20));
  bf16* wo_t   = (bf16*)(ws + (14u << 20));
  bf16* q_ws   = (bf16*)(ws + (16u << 20));
  bf16* k_ws   = (bf16*)(ws + (24u << 20));
  bf16* v_ws   = (bf16*)(ws + (32u << 20));
  bf16* ctx_ws = (bf16*)(ws + (40u << 20));

  const int nHid = BB * SS * DMm;            // 4 Mi elements
  const int nW   = DMm * DMm;                // 1 Mi elements

  k_cvt_bf16<<<(nHid + 255) / 256, 256, 0, stream>>>(hid, hid_bf, nHid);
  k_cvt_t<<<nW / 256, 256, 0, stream>>>(Wq, wq_t);
  k_cvt_t<<<nW / 256, 256, 0, stream>>>(Wk, wk_t);
  k_cvt_t<<<nW / 256, 256, 0, stream>>>(Wv, wv_t);
  k_cvt_t<<<nW / 256, 256, 0, stream>>>(Wo, wo_t);

  k_qkv_rope<<<dim3(256, 6), 256, 0, stream>>>(hid_bf, wq_t, wk_t, wv_t, pos,
                                               q_ws, k_ws, v_ws);
  k_attn<<<dim3(32, 16), 256, 0, stream>>>(q_ws, k_ws, v_ws, mask, ctx_ws);
  k_oproj<<<dim3(256, 8), 256, 0, stream>>>(ctx_ws, wo_t, (float*)d_out);
}
